// GlobalExpert_39728447488795
// MI455X (gfx1250) — matrix operand order repaired and checked by reading, not compiled, not run on hardware
//
#include <hip/hip_runtime.h>
#include <math.h>

#define D_MODEL 1024
#define D_STATE 16
#define D_CONV 4
#define D_INNER 2048
#define DT_RANK 64
#define BATCH 2
#define SEQ 2048
#define NROWS (BATCH * SEQ)        // 4096

#define TILE_M 128
#define TILE_N 128
#define TILE_K 32
#define LDSS 40                    // padded LDS row stride (bf16 elems), multiple of 8

typedef __attribute__((ext_vector_type(8)))  __bf16 v8bf;
typedef __attribute__((ext_vector_type(16))) __bf16 v16bf;
typedef __attribute__((ext_vector_type(8)))  float  v8f;
typedef __attribute__((ext_vector_type(4)))  int    v4i;

__device__ __forceinline__ unsigned short f2bf(float f) {
    unsigned int u = __float_as_uint(f);
    u += 0x7fffu + ((u >> 16) & 1u);          // round-to-nearest-even
    return (unsigned short)(u >> 16);
}

__device__ __forceinline__ float silu_f(float x) {
    return x * (1.0f / (1.0f + __expf(-x)));
}

// 16-byte global -> LDS copy; async path (ASYNCcnt) on gfx1250 when available.
__device__ __forceinline__ void copy16_g2l(const unsigned short* g, unsigned short* l) {
#if defined(__AMDGCN__) && __has_builtin(__builtin_amdgcn_global_load_async_to_lds_b128)
    __builtin_amdgcn_global_load_async_to_lds_b128(
        (__attribute__((address_space(1))) v4i*)g,
        (__attribute__((address_space(3))) v4i*)l, 0, 0);
#else
    *(uint4*)l = *(const uint4*)g;
#endif
}

__device__ __forceinline__ void async_wait_all() {
#if defined(__gfx1250__)
    asm volatile("s_wait_asynccnt 0" ::: "memory");
#endif
}

// C[M,N](f32) = A[M,K](bf16, row stride lda elems) * W[N,K](bf16)^T via WMMA.
// Double-buffered: async-DMA stage of tile k+1 overlaps WMMA compute on tile k.
__global__ __launch_bounds__(256) void gemm_bf16_wmma(
    const unsigned short* __restrict__ A, int lda,
    const unsigned short* __restrict__ W,
    float* __restrict__ C, int M, int N, int K)
{
    __shared__ unsigned short sA[2][TILE_M][LDSS];
    __shared__ unsigned short sB[2][TILE_N][LDSS];

    const int tid   = threadIdx.x;
    const int lane  = tid & 31;
    const int wave  = tid >> 5;        // 0..7
    const int waveM = wave >> 1;       // 0..3  -> 32-row strip
    const int waveN = wave & 1;        // 0..1  -> 64-col strip
    const int bM = blockIdx.y * TILE_M;
    const int bN = blockIdx.x * TILE_N;

    const int r    = lane & 15;
    const int half = lane >> 4;

    // each thread stages 2 x 16B chunks of A and of B per tile
    const int srow = tid >> 2;         // 0..63 (j adds 64)
    const int sc   = tid & 3;

    auto stage = [&](int buf, int k0) {
        #pragma unroll
        for (int j = 0; j < 2; ++j) {
            int row = srow + j * 64;
            copy16_g2l(A + (size_t)(bM + row) * lda + k0 + sc * 8,
                       &sA[buf][row][sc * 8]);
        }
        #pragma unroll
        for (int j = 0; j < 2; ++j) {
            int row = srow + j * 64;
            if (bN + row < N) {
                copy16_g2l(W + (size_t)(bN + row) * K + k0 + sc * 8,
                           &sB[buf][row][sc * 8]);
            } else {
                *(uint4*)&sB[buf][row][sc * 8] = make_uint4(0u, 0u, 0u, 0u);
            }
        }
    };

    v8f acc[2][4];
    #pragma unroll
    for (int mi = 0; mi < 2; ++mi)
        #pragma unroll
        for (int ni = 0; ni < 4; ++ni)
            #pragma unroll
            for (int v = 0; v < 8; ++v)
                acc[mi][ni][v] = 0.0f;

    stage(0, 0);
    async_wait_all();
    __syncthreads();

    int buf = 0;
    for (int k0 = 0; k0 < K; k0 += TILE_K) {
        // prefetch next tile into the other buffer (async, not yet waited)
        if (k0 + TILE_K < K) stage(buf ^ 1, k0 + TILE_K);

        // ---- A fragments: lane(r)=M row; K groups {0..7,16..23}+8*half ----
        v16bf afr[2];
        #pragma unroll
        for (int mi = 0; mi < 2; ++mi) {
            int row = waveM * 32 + mi * 16 + r;
            v8bf lo = *(const v8bf*)&sA[buf][row][half * 8];
            v8bf hi = *(const v8bf*)&sA[buf][row][16 + half * 8];
            afr[mi] = __builtin_shufflevector(lo, hi,
                        0,1,2,3,4,5,6,7,8,9,10,11,12,13,14,15);
        }
        // ---- B fragments + WMMA: lane(r)=N col; K groups {0..7,16..23}+8*half ----
        #pragma unroll
        for (int ni = 0; ni < 4; ++ni) {
            int col = waveN * 64 + ni * 16 + r;
            v8bf lo = *(const v8bf*)&sB[buf][col][half * 8];
            v8bf hi = *(const v8bf*)&sB[buf][col][16 + half * 8];
            v16bf bfr = __builtin_shufflevector(lo, hi,
                          0,1,2,3,4,5,6,7,8,9,10,11,12,13,14,15);
            #pragma unroll
            for (int mi = 0; mi < 2; ++mi)
                acc[mi][ni] = __builtin_amdgcn_wmma_f32_16x16x32_bf16(
                    false, afr[mi], false, bfr, (short)0, acc[mi][ni], false, false);
        }

        // own async copies done -> barrier publishes next buffer to all waves
        async_wait_all();
        __syncthreads();
        buf ^= 1;
    }

    // ---- store: D layout = lane(r) col, VGPR v -> M = v + 8*half ----
    #pragma unroll
    for (int mi = 0; mi < 2; ++mi)
        #pragma unroll
        for (int ni = 0; ni < 4; ++ni) {
            int colg = bN + waveN * 64 + ni * 16 + r;
            if (colg < N) {
                #pragma unroll
                for (int v = 0; v < 8; ++v) {
                    int rowg = bM + waveM * 32 + mi * 16 + v + 8 * half;
                    C[(size_t)rowg * N + colg] = acc[mi][ni][v];
                }
            }
        }
}

// f32 -> bf16 pack, 8 elems per thread
__global__ __launch_bounds__(256) void f32_to_bf16_kernel(
    const float* __restrict__ in, unsigned short* __restrict__ out, int n8)
{
    int i = blockIdx.x * blockDim.x + threadIdx.x;
    if (i >= n8) return;
    const float4 a = ((const float4*)in)[i * 2];
    const float4 b = ((const float4*)in)[i * 2 + 1];
    uint4 o;
    o.x = (unsigned int)f2bf(a.x) | ((unsigned int)f2bf(a.y) << 16);
    o.y = (unsigned int)f2bf(a.z) | ((unsigned int)f2bf(a.w) << 16);
    o.z = (unsigned int)f2bf(b.x) | ((unsigned int)f2bf(b.y) << 16);
    o.w = (unsigned int)f2bf(b.z) | ((unsigned int)f2bf(b.w) << 16);
    ((uint4*)out)[i] = o;
}

// causal depthwise conv1d (+bias, SiLU); writes f32 (for scan/gate) and bf16 (for x_proj GEMM)
__global__ __launch_bounds__(256) void conv_silu_kernel(
    const float* __restrict__ xz, const float* __restrict__ conv_w,
    const float* __restrict__ conv_b, float* __restrict__ xc,
    unsigned short* __restrict__ xc_bf)
{
    int i = blockIdx.x * blockDim.x + threadIdx.x;
    if (i >= NROWS * D_INNER) return;
    int d  = i & (D_INNER - 1);
    int ml = i >> 11;                 // row index (b*SEQ + l)
    int l  = ml & (SEQ - 1);
    int b  = ml >> 11;
    float acc = conv_b[d];
    #pragma unroll
    for (int j = 0; j < D_CONV; ++j) {
        int ll = l - (D_CONV - 1) + j;
        if (ll >= 0)
            acc += conv_w[d * D_CONV + j] *
                   xz[(size_t)(b * SEQ + ll) * (2 * D_INNER) + d];
    }
    float s = silu_f(acc);
    xc[i] = s;
    xc_bf[i] = f2bf(s);
}

// dt = softplus(dt_lin + bias), in place
__global__ __launch_bounds__(256) void dt_bias_softplus_kernel(
    float* __restrict__ dt, const float* __restrict__ dt_proj_b)
{
    int i = blockIdx.x * blockDim.x + threadIdx.x;
    if (i >= NROWS * D_INNER) return;
    int d = i & (D_INNER - 1);
    float x = dt[i] + dt_proj_b[d];
    dt[i] = (x > 20.0f) ? x : log1pf(__expf(x));
}

// selective scan: one thread per (batch, channel)
__global__ __launch_bounds__(256) void scan_kernel(
    const float* __restrict__ dt, const float* __restrict__ xc,
    const float* __restrict__ xdbl, const float* __restrict__ A_log,
    float* __restrict__ ys)
{
    int tid = blockIdx.x * blockDim.x + threadIdx.x;
    if (tid >= BATCH * D_INNER) return;
    int b = tid >> 11;
    int d = tid & (D_INNER - 1);
    float Acoef[D_STATE], h[D_STATE];
    #pragma unroll
    for (int s = 0; s < D_STATE; ++s) {
        Acoef[s] = -__expf(A_log[d * D_STATE + s]);
        h[s] = 0.0f;
    }
    const size_t rowbase = (size_t)b * SEQ;
    for (int l = 0; l < SEQ; ++l) {
        size_t m   = rowbase + l;
        float dtt  = dt[m * D_INNER + d];
        float xt   = xc[m * D_INNER + d];
        float dx   = dtt * xt;
        const float* bc = xdbl + m * (DT_RANK + 2 * D_STATE) + DT_RANK;
        float y = 0.0f;
        #pragma unroll
        for (int s = 0; s < D_STATE; ++s) {
            float dA = __expf(dtt * Acoef[s]);
            h[s] = dA * h[s] + dx * bc[s];
            y += h[s] * bc[D_STATE + s];
        }
        ys[m * D_INNER + d] = y;
    }
}

// yg_bf = bf16( (ys + xc*D) * silu(z) );  z = xz[:, DI:2*DI]
__global__ __launch_bounds__(256) void gate_kernel(
    const float* __restrict__ ys, const float* __restrict__ xc,
    const float* __restrict__ xz, const float* __restrict__ D_param,
    unsigned short* __restrict__ yg_bf)
{
    int i = blockIdx.x * blockDim.x + threadIdx.x;
    if (i >= NROWS * D_INNER) return;
    int d = i & (D_INNER - 1);
    int m = i >> 11;
    float z = xz[(size_t)m * (2 * D_INNER) + D_INNER + d];
    yg_bf[i] = f2bf((ys[i] + xc[i] * D_param[d]) * silu_f(z));
}

extern "C" void kernel_launch(void* const* d_in, const int* in_sizes, int n_in,
                              void* d_out, int out_size, void* d_ws, size_t ws_size,
                              hipStream_t stream) {
    (void)in_sizes; (void)n_in; (void)out_size; (void)ws_size;
    const float* x         = (const float*)d_in[0];
    const float* in_proj_w = (const float*)d_in[1];
    const float* conv_w    = (const float*)d_in[2];
    const float* conv_b    = (const float*)d_in[3];
    const float* x_proj_w  = (const float*)d_in[4];
    const float* dt_proj_w = (const float*)d_in[5];
    const float* dt_proj_b = (const float*)d_in[6];
    const float* A_log     = (const float*)d_in[7];
    const float* D_param   = (const float*)d_in[8];
    const float* out_proj_w= (const float*)d_in[9];
    float* out = (float*)d_out;

    char* ws = (char*)d_ws;
    // ---- f32 buffers ----
    float* xz    = (float*)(ws);                       // 4096*4096*4 = 64 MB
    float* xc    = (float*)(ws + (size_t)67108864);    // 4096*2048*4 = 32 MB
    float* x_dbl = (float*)(ws + (size_t)100663296);   // 4096*96*4
    float* dt    = (float*)(ws + (size_t)102236160);   // 4096*2048*4 = 32 MB
    float* ys    = (float*)(ws + (size_t)135790592);   // 4096*2048*4 = 32 MB
    // ---- bf16 staging buffers, time-multiplexed into dead regions ----
    // Phase 1 (before GEMM1) lives in the ys region (ys written only at scan):
    unsigned short* x_bf    = (unsigned short*)(ws + (size_t)135790592);            // 8 MB
    unsigned short* w_in_bf = (unsigned short*)(ws + (size_t)135790592 + 8388608);  // 8 MB
    // Phase 2 (after GEMM1, before scan) also in ys region:
    unsigned short* xc_bf   = (unsigned short*)(ws + (size_t)135790592);            // 16 MB
    unsigned short* w_xp_bf = (unsigned short*)(ws + (size_t)135790592 + 16777216); // 384 KB
    unsigned short* xdbl_bf = (unsigned short*)(ws + (size_t)135790592 + 17170432); // 768 KB
    unsigned short* w_dt_bf = (unsigned short*)(ws + (size_t)135790592 + 17956864); // 256 KB
    // Phase 3 (after scan) in dt region (dt f32 dead after scan):
    unsigned short* yg_bf   = (unsigned short*)(ws + (size_t)102236160);            // 16 MB
    unsigned short* w_out_bf= (unsigned short*)(ws + (size_t)102236160 + 16777216); // 4 MB

    dim3 blk(256);
    #define CVT(src, dst, nelem) \
        f32_to_bf16_kernel<<<(((nelem) / 8) + 255) / 256, blk, 0, stream>>>( \
            src, dst, (nelem) / 8)

    // ---- Phase 1: convert x + in_proj_w, then in_proj GEMM ----
    CVT(x, x_bf, NROWS * D_MODEL);
    CVT(in_proj_w, w_in_bf, 2 * D_INNER * D_MODEL);
    {
        dim3 grid((2 * D_INNER) / TILE_N, NROWS / TILE_M);
        gemm_bf16_wmma<<<grid, blk, 0, stream>>>(x_bf, D_MODEL, w_in_bf, xz,
                                                 NROWS, 2 * D_INNER, D_MODEL);
    }
    // ---- Phase 2: conv (emits xc f32 + bf16), x_proj, dt_proj ----
    conv_silu_kernel<<<(NROWS * D_INNER + 255) / 256, blk, 0, stream>>>(
        xz, conv_w, conv_b, xc, xc_bf);
    CVT(x_proj_w, w_xp_bf, (DT_RANK + 2 * D_STATE) * D_INNER);
    {
        dim3 grid((DT_RANK + 2 * D_STATE + TILE_N - 1) / TILE_N, NROWS / TILE_M);
        gemm_bf16_wmma<<<grid, blk, 0, stream>>>(xc_bf, D_INNER, w_xp_bf, x_dbl,
                                                 NROWS, DT_RANK + 2 * D_STATE, D_INNER);
    }
    CVT(x_dbl, xdbl_bf, NROWS * (DT_RANK + 2 * D_STATE));
    CVT(dt_proj_w, w_dt_bf, D_INNER * DT_RANK);
    {
        dim3 grid(D_INNER / TILE_N, NROWS / TILE_M);
        gemm_bf16_wmma<<<grid, blk, 0, stream>>>(xdbl_bf, DT_RANK + 2 * D_STATE,
                                                 w_dt_bf, dt,
                                                 NROWS, D_INNER, DT_RANK);
    }
    dt_bias_softplus_kernel<<<(NROWS * D_INNER + 255) / 256, blk, 0, stream>>>(
        dt, dt_proj_b);
    // ---- scan (reads dt/xc/x_dbl f32, writes ys; frees all phase-2 bf16) ----
    scan_kernel<<<(BATCH * D_INNER + 255) / 256, blk, 0, stream>>>(
        dt, xc, x_dbl, A_log, ys);
    // ---- Phase 3: gate -> yg_bf (into dead dt region), out_proj GEMM ----
    CVT(out_proj_w, w_out_bf, D_MODEL * D_INNER);   // after scan: dt region is dead
    gate_kernel<<<(NROWS * D_INNER + 255) / 256, blk, 0, stream>>>(
        ys, xc, xz, D_param, yg_bf);
    {
        dim3 grid(D_MODEL / TILE_N, NROWS / TILE_M);
        gemm_bf16_wmma<<<grid, blk, 0, stream>>>(yg_bf, D_INNER, w_out_bf, out,
                                                 NROWS, D_MODEL, D_INNER);
    }
    #undef CVT
}
